// BondEncoder_34780645163257
// MI455X (gfx1250) — matrix operand order repaired and checked by reading, not compiled, not run on hardware
//
#include <hip/hip_runtime.h>
#include <hip/hip_bf16.h>

// ---------------------------------------------------------------------------
// Types for CDNA5 WMMA (wave32): v_wmma_f32_16x16x32_bf16
// ---------------------------------------------------------------------------
typedef __attribute__((ext_vector_type(16))) __bf16 v16bf;
typedef __attribute__((ext_vector_type(8)))  float  v8f;

struct alignas(16) U128 { unsigned int w[4]; };
union FragAB { U128 q[2]; v16bf v; };

__device__ __forceinline__ unsigned short f2bf(float f) {
    unsigned int u = __float_as_uint(f);
    unsigned int r = u + 0x7FFFu + ((u >> 16) & 1u);   // round-to-nearest-even
    return (unsigned short)(r >> 16);
}

#define WMMA_BF16(A, Bv, C) \
    __builtin_amdgcn_wmma_f32_16x16x32_bf16(false, (A), false, (Bv), (short)0, (C), false, false)

// Problem constants
#define BATCH 256
#define NA    128   // N_ATOMS == T
#define INF   128   // IN_F
#define HID   256   // H
#define G3    768   // 3*H
#define OUTF  256
#define KFC   1536  // H*MAX_VAL

// ---------------------------------------------------------------------------
// Kernel 0: f32 -> bf16 conversion (grid-stride)
// ---------------------------------------------------------------------------
__global__ void cvt_bf16_kernel(const float* __restrict__ src,
                                unsigned short* __restrict__ dst, int n) {
    int i = blockIdx.x * blockDim.x + threadIdx.x;
    int stride = gridDim.x * blockDim.x;
    for (; i < n; i += stride) dst[i] = f2bf(src[i]);
}

// ---------------------------------------------------------------------------
// Kernel 1: fused GRU. 16 blocks x 512 threads (16 waves).
// Each block owns 16 batch rows for all 128 timesteps; no inter-block sync.
// Wave w owns hidden columns c = 16w..16w+15 and computes the r, z and n gate
// tiles for those columns as 4 independent interleaved WMMA chains
// (r:12, z:12, hn:8, xn:4 WMMAs). All accumulators and the persistent f32
// hidden state share the same C-matrix register layout, so the whole GRU
// nonlinearity runs in registers; only the bf16 h tile (8.4KB LDS) is shared.
// Weights are re-streamed from L2 every step. LICM is blocked by an opaque
// 32-bit offset (not an opaque pointer: that would erase the global address
// space and demote the loads to FLAT, entangling them with the LDS counter).
// ---------------------------------------------------------------------------
__launch_bounds__(512, 1)
__global__ void gru_kernel(const unsigned short* __restrict__ x_bf,   // [B][128][128] bf16
                           const unsigned short* __restrict__ wih_bf, // [768][128]  bf16
                           const unsigned short* __restrict__ whh_bf, // [768][256]  bf16
                           const float* __restrict__ b_ih,            // [768]
                           const float* __restrict__ b_hh,            // [768]
                           unsigned short* __restrict__ rnn_bf)       // [B][128][256] bf16
{
    __shared__ unsigned short s_h[16][264];    // bf16 hidden state, padded rows

    const int tid   = threadIdx.x;
    const int lane  = tid & 31;
    const int wave  = tid >> 5;      // 0..15
    const int b0    = blockIdx.x * 16;
    const int mrow  = lane & 15;     // A-row / B-col selector
    const int khalf = lane >> 4;     // lane half selects K sub-range

    const int c = wave * 16 + mrow;  // owned hidden column 0..255 (C-layout N index)

    // zero bf16 h
    for (int i = tid; i < 16 * 264; i += 512) (&s_h[0][0])[i] = 0;
    __syncthreads();

    // hoisted biases (r/z: fused x+h bias; n: keep sides separate)
    const float bias_r  = b_ih[c]       + b_hh[c];
    const float bias_z  = b_ih[c + 256] + b_hh[c + 256];
    const float bias_xn = b_ih[c + 512];
    const float bias_hn = b_hh[c + 512];

    // weight row pointers (row = gate column, khalf picks K sub-block)
    const unsigned short* whh_r = whh_bf + (size_t)(c      ) * HID + khalf * 16;
    const unsigned short* whh_z = whh_bf + (size_t)(c + 256) * HID + khalf * 16;
    const unsigned short* whh_n = whh_bf + (size_t)(c + 512) * HID + khalf * 16;
    const unsigned short* wih_r = wih_bf + (size_t)(c      ) * INF + khalf * 16;
    const unsigned short* wih_z = wih_bf + (size_t)(c + 256) * INF + khalf * 16;
    const unsigned short* wih_n = wih_bf + (size_t)(c + 512) * INF + khalf * 16;

    const int m0 = khalf * 8;        // C-layout row base for this lane

    float hreg[8];                   // f32 master hidden state, C layout
#pragma unroll
    for (int r = 0; r < 8; ++r) hreg[r] = 0.f;

    for (int t = 0; t < NA; ++t) {
        // ---- hidden-state A fragments from LDS ----
        FragAB ah[8];
#pragma unroll
        for (int kt = 0; kt < 8; ++kt) {
            const unsigned short* p = &s_h[mrow][kt * 32 + khalf * 16];
            ah[kt].q[0] = *(const U128*)(p);
            ah[kt].q[1] = *(const U128*)(p + 8);
        }
        __syncthreads();             // everyone done reading s_h

        // Opaque zero offset: weight addresses become loop-defined (no LICM,
        // no 288-VGPR hoist+spill) while pointer provenance stays global
        // (loads select global_load_b128, tracked by LOADcnt only).
        unsigned int zof = 0;
        asm volatile("" : "+v"(zof));
        const unsigned short* pwhh_r = whh_r + zof;
        const unsigned short* pwhh_z = whh_z + zof;
        const unsigned short* pwhh_n = whh_n + zof;
        const unsigned short* pwih_r = wih_r + zof;
        const unsigned short* pwih_z = wih_z + zof;
        const unsigned short* pwih_n = wih_n + zof;

        v8f acc_r, acc_z, acc_xn, acc_hn;
#pragma unroll
        for (int r = 0; r < 8; ++r) {
            acc_r[r]  = bias_r;
            acc_z[r]  = bias_z;
            acc_xn[r] = bias_xn;
            acc_hn[r] = bias_hn;
        }

        // hidden K = 256 : three independent chains per k-tile
#pragma unroll
        for (int kt = 0; kt < 8; ++kt) {
            const int o = kt * 32;
            FragAB br, bz, bn;
            br.q[0] = *(const U128*)(pwhh_r + o); br.q[1] = *(const U128*)(pwhh_r + o + 8);
            bz.q[0] = *(const U128*)(pwhh_z + o); bz.q[1] = *(const U128*)(pwhh_z + o + 8);
            bn.q[0] = *(const U128*)(pwhh_n + o); bn.q[1] = *(const U128*)(pwhh_n + o + 8);
            acc_r  = WMMA_BF16(ah[kt].v, br.v, acc_r);
            acc_z  = WMMA_BF16(ah[kt].v, bz.v, acc_z);
            acc_hn = WMMA_BF16(ah[kt].v, bn.v, acc_hn);
        }
        // input K = 128 : A fragment loaded per k-tile (global, L2 resident)
        const unsigned short* xrow =
            x_bf + ((size_t)(b0 + mrow) * NA + t) * INF + khalf * 16;
#pragma unroll
        for (int kt = 0; kt < 4; ++kt) {
            const int o = kt * 32;
            FragAB axk;
            axk.q[0] = *(const U128*)(xrow + o);
            axk.q[1] = *(const U128*)(xrow + o + 8);
            FragAB br, bz, bn;
            br.q[0] = *(const U128*)(pwih_r + o); br.q[1] = *(const U128*)(pwih_r + o + 8);
            bz.q[0] = *(const U128*)(pwih_z + o); bz.q[1] = *(const U128*)(pwih_z + o + 8);
            bn.q[0] = *(const U128*)(pwih_n + o); bn.q[1] = *(const U128*)(pwih_n + o + 8);
            acc_r  = WMMA_BF16(axk.v, br.v, acc_r);
            acc_z  = WMMA_BF16(axk.v, bz.v, acc_z);
            acc_xn = WMMA_BF16(axk.v, bn.v, acc_xn);
        }

        // ---- GRU nonlinearity fully in registers (layouts all line up) ----
        unsigned short hb[8];
#pragma unroll
        for (int r = 0; r < 8; ++r) {
            const float rv = 1.f / (1.f + __expf(-acc_r[r]));
            const float zv = 1.f / (1.f + __expf(-acc_z[r]));
            const float nv = tanhf(acc_xn[r] + rv * acc_hn[r]);
            const float hnew = (1.f - zv) * nv + zv * hreg[r];
            hreg[r] = hnew;
            hb[r] = f2bf(hnew);
        }
        // publish h_t (LDS for next step's A-fragments) + rnn output (global)
#pragma unroll
        for (int r = 0; r < 8; ++r) {
            const int m = m0 + r;
            s_h[m][c] = hb[r];
            rnn_bf[((size_t)(b0 + m) * NA + t) * HID + c] = hb[r];
        }
        __syncthreads();             // s_h complete for next step
    }
}

// ---------------------------------------------------------------------------
// Kernel 2: gather + FC GEMM + LeakyReLU.
// out[row, o] = leaky( sum_{v,c} rnn[b, bonded[b,a,v], c] * W_fc[o, v*256+c] + b_fc[o] )
// 512 blocks x 256 threads = 4096 waves; wave = one 16-row M-tile x 8 N-tiles
// (8 independent WMMA chains). Gather folded into A-fragment addresses.
// ---------------------------------------------------------------------------
__launch_bounds__(256, 2)
__global__ void fc_kernel(const unsigned short* __restrict__ rnn_bf, // [B*128][256] bf16
                          const int* __restrict__ bonded,            // [B][128][6]
                          const unsigned short* __restrict__ wfc_bf, // [256][1536] bf16
                          const float* __restrict__ b_fc,            // [256]
                          float* __restrict__ out)                   // [B*128][256] f32
{
    const int lane  = threadIdx.x & 31;
    const int wave  = threadIdx.x >> 5;
    const int wgid  = blockIdx.x * 8 + wave;   // 0..4095
    const int mtile = wgid >> 1;               // 0..2047
    const int ngrp  = wgid & 1;                // low/high 8 OUT_F tiles
    const int mrow  = lane & 15;
    const int khalf = lane >> 4;

    const int grow = mtile * 16 + mrow;        // global (b,a) row 0..32767
    const int b = grow >> 7, a = grow & 127;
    const int* bp = bonded + ((size_t)b * NA + a) * 6;

    const unsigned short* arow0 = rnn_bf + ((size_t)b * NA + bp[0]) * HID;
    const unsigned short* arow1 = rnn_bf + ((size_t)b * NA + bp[1]) * HID;
    const unsigned short* arow2 = rnn_bf + ((size_t)b * NA + bp[2]) * HID;
    const unsigned short* arow3 = rnn_bf + ((size_t)b * NA + bp[3]) * HID;
    const unsigned short* arow4 = rnn_bf + ((size_t)b * NA + bp[4]) * HID;
    const unsigned short* arow5 = rnn_bf + ((size_t)b * NA + bp[5]) * HID;
    const unsigned short* arows[6] = { arow0, arow1, arow2, arow3, arow4, arow5 };

    v8f acc[8];
#pragma unroll
    for (int nt = 0; nt < 8; ++nt) {
        const float bias = b_fc[(ngrp * 8 + nt) * 16 + mrow];
#pragma unroll
        for (int r = 0; r < 8; ++r) acc[nt][r] = bias;
    }

#pragma unroll
    for (int v = 0; v < 6; ++v) {              // unrolled: arows[v] stays in regs
        const unsigned short* ar = arows[v];
#pragma unroll 2
        for (int kt2 = 0; kt2 < 8; ++kt2) {
            const int kt = v * 8 + kt2;        // global k-tile 0..47
            const int c0 = kt2 * 32 + khalf * 16;
            FragAB af;
            af.q[0] = *(const U128*)(ar + c0);
            af.q[1] = *(const U128*)(ar + c0 + 8);
#pragma unroll
            for (int nt = 0; nt < 8; ++nt) {
                const int o = (ngrp * 8 + nt) * 16 + mrow;
                FragAB bw;
                const unsigned short* p = wfc_bf + (size_t)o * KFC + kt * 32 + khalf * 16;
                bw.q[0] = *(const U128*)(p);
                bw.q[1] = *(const U128*)(p + 8);
                acc[nt] = WMMA_BF16(af.v, bw.v, acc[nt]);
            }
        }
    }

#pragma unroll
    for (int nt = 0; nt < 8; ++nt) {
        const int o = (ngrp * 8 + nt) * 16 + mrow;
#pragma unroll
        for (int r = 0; r < 8; ++r) {
            const int m = r + khalf * 8;
            float vvl = acc[nt][r];
            vvl = (vvl >= 0.f) ? vvl : 0.1f * vvl;
            out[((size_t)mtile * 16 + m) * OUTF + o] = vvl;
        }
    }
}

// ---------------------------------------------------------------------------
// Launch
// ---------------------------------------------------------------------------
extern "C" void kernel_launch(void* const* d_in, const int* in_sizes, int n_in,
                              void* d_out, int out_size, void* d_ws, size_t ws_size,
                              hipStream_t stream) {
    const float* x      = (const float*)d_in[0];
    const int*   bonded = (const int*)d_in[1];
    const float* W_ih   = (const float*)d_in[2];
    const float* W_hh   = (const float*)d_in[3];
    const float* b_ih   = (const float*)d_in[4];
    const float* b_hh   = (const float*)d_in[5];
    const float* W_fc   = (const float*)d_in[6];
    const float* b_fc   = (const float*)d_in[7];
    float* out = (float*)d_out;

    // workspace layout (bf16 as ushort), 256B aligned
    char* ws = (char*)d_ws;
    const size_t nX   = (size_t)BATCH * NA * INF;   // 4,194,304
    const size_t nWih = (size_t)G3 * INF;           //    98,304
    const size_t nWhh = (size_t)G3 * HID;           //   196,608
    const size_t nWfc = (size_t)OUTF * KFC;         //   393,216

    size_t off = 0;
    unsigned short* x_bf   = (unsigned short*)(ws + off); off += nX * 2;   off = (off + 255) & ~(size_t)255;
    unsigned short* wih_bf = (unsigned short*)(ws + off); off += nWih * 2; off = (off + 255) & ~(size_t)255;
    unsigned short* whh_bf = (unsigned short*)(ws + off); off += nWhh * 2; off = (off + 255) & ~(size_t)255;
    unsigned short* wfc_bf = (unsigned short*)(ws + off); off += nWfc * 2; off = (off + 255) & ~(size_t)255;
    unsigned short* rnn_bf = (unsigned short*)(ws + off);

    cvt_bf16_kernel<<<2048, 256, 0, stream>>>(x,    x_bf,   (int)nX);
    cvt_bf16_kernel<<<96,   256, 0, stream>>>(W_ih, wih_bf, (int)nWih);
    cvt_bf16_kernel<<<192,  256, 0, stream>>>(W_hh, whh_bf, (int)nWhh);
    cvt_bf16_kernel<<<384,  256, 0, stream>>>(W_fc, wfc_bf, (int)nWfc);

    gru_kernel<<<16, 512, 0, stream>>>(x_bf, wih_bf, whh_bf, b_ih, b_hh, rnn_bf);

    fc_kernel<<<512, 256, 0, stream>>>(rnn_bf, bonded, wfc_bf, b_fc, out);
}
